// MultiScanMamba_46342697123956
// MI455X (gfx1250) — compile-verified, matrix operand order checked by reading, not run on hardware
//
#include <hip/hip_runtime.h>

// ---------------------------------------------------------------------------
// MultiScan Mamba for gfx1250 (MI455X).
//  - GEMMs:   v_wmma_f32_16x16x32_bf16, 2x2 tiles/wave, double-buffered frags
//  - scan:    fp32 VALU recurrence, B/C staged through LDS in chunks
//  - sort:    hilbert / morton codes + in-LDS bitonic argsort
// ---------------------------------------------------------------------------

#define B_   2
#define N_   2048
#define T_   (B_ * N_)     // 4096 tokens
#define D_   256
#define DI_  512
#define DS_  16
#define XD_  48            // dt_rank + 2*d_state (logical)
#define XDP_ 64            // padded x_dbl stride (multiple of 64 for GEMM tiling)
#define CD_  512

typedef __bf16 bhalf;
typedef __attribute__((ext_vector_type(16))) __bf16 v16bf;
typedef __attribute__((ext_vector_type(8)))  __bf16 v8bf;
typedef __attribute__((ext_vector_type(8)))  float  v8f;

// ---------------------------------------------------------------------------
// fp32 -> bf16 conversion with optional zero padding (row-major append)
// ---------------------------------------------------------------------------
__global__ void k_f32_to_bf16_pad(const float* __restrict__ s, bhalf* __restrict__ d,
                                  int n_src, int n_dst) {
  int i = blockIdx.x * blockDim.x + threadIdx.x;
  if (i < n_dst) d[i] = (bhalf)((i < n_src) ? s[i] : 0.0f);
}

// ---------------------------------------------------------------------------
// Hilbert (Skilling) encode, 16 bits x 3 dims -> 48-bit code
// ---------------------------------------------------------------------------
__device__ unsigned long long hilbert48(unsigned int v0, unsigned int v1, unsigned int v2) {
  const int nb = 16;
  unsigned int g[3] = {v0, v1, v2};
  for (int bit = 0; bit < nb; ++bit) {
    int p = nb - 1 - bit;
    unsigned int low = (p == 0) ? 0u : ((1u << p) - 1u);
    for (int dim = 0; dim < 3; ++dim) {
      unsigned int mask  = (g[dim] >> p) & 1u;
      unsigned int mfill = mask ? low : 0u;
      unsigned int g0    = (g[0] & low) ^ mfill;
      if (dim == 0) {
        g[0] = (g[0] & ~low) | g0;
      } else {
        unsigned int gd   = g[dim] & low;
        unsigned int flip = mask ? 0u : (g0 ^ gd);
        g[dim] = (g[dim] & ~low) | (gd ^ flip);
        g[0]   = (g[0]  & ~low) | (g0 ^ flip);
      }
    }
  }
  unsigned long long code = 0; unsigned int acc = 0;
  for (int bit = 0; bit < nb; ++bit)
    for (int dim = 0; dim < 3; ++dim) {
      acc ^= (g[dim] >> (nb - 1 - bit)) & 1u;
      code = (code << 1) | (unsigned long long)acc;
    }
  return code;
}

__device__ unsigned int spread10(unsigned int v) {
  v &= 1023u;
  v = (v | (v << 16)) & 50331903u;
  v = (v | (v << 8))  & 50393103u;
  v = (v | (v << 4))  & 51130563u;
  v = (v | (v << 2))  & 153391689u;
  return v;
}

// One block per (batch, order): order 0 = hilbert(16b), order 1 = z-order(10b).
__global__ __launch_bounds__(1024) void k_sort(const float* __restrict__ xyz,
                                               int* __restrict__ sidx,
                                               int* __restrict__ uidx) {
  const int order = blockIdx.x & 1;
  const int b     = blockIdx.x >> 1;
  const int tid   = threadIdx.x;
  __shared__ unsigned long long key[N_];
  __shared__ float sred[1024];
  __shared__ float mnmx[6];

  // --- min/max over N per coordinate ---
  for (int c = 0; c < 3; ++c) {
    float a0 = xyz[((size_t)(b * N_ + tid)       ) * 3 + c];
    float a1 = xyz[((size_t)(b * N_ + tid + 1024)) * 3 + c];
    sred[tid] = fminf(a0, a1); __syncthreads();
    for (int o = 512; o > 0; o >>= 1) { if (tid < o) sred[tid] = fminf(sred[tid], sred[tid + o]); __syncthreads(); }
    if (tid == 0) mnmx[c] = sred[0];
    __syncthreads();
    sred[tid] = fmaxf(a0, a1); __syncthreads();
    for (int o = 512; o > 0; o >>= 1) { if (tid < o) sred[tid] = fmaxf(sred[tid], sred[tid + o]); __syncthreads(); }
    if (tid == 0) mnmx[3 + c] = sred[0];
    __syncthreads();
  }

  // --- encode ---
  for (int e = tid; e < N_; e += 1024) {
    unsigned int q[3];
    for (int c = 0; c < 3; ++c) {
      float v  = xyz[((size_t)(b * N_ + e)) * 3 + c];
      float nm = (v - mnmx[c]) / (mnmx[3 + c] - mnmx[c] + 1e-8f);
      if (order == 0) q[c] = (unsigned int)(nm * 0.999f * 65535.0f);
      else            q[c] = (unsigned int)(nm * 1023.0f);
    }
    unsigned long long code;
    if (order == 0) code = hilbert48(q[0], q[1], q[2]);
    else code = (unsigned long long)(spread10(q[0]) | (spread10(q[1]) << 1) | (spread10(q[2]) << 2));
    key[e] = (code << 11) | (unsigned long long)e;
  }

  // --- bitonic sort (ascending) ---
  for (unsigned int k = 2; k <= N_; k <<= 1)
    for (unsigned int j = k >> 1; j > 0; j >>= 1) {
      __syncthreads();
      for (int e = tid; e < N_; e += 1024) {
        int p = e ^ (int)j;
        if (p > e) {
          bool up = ((e & k) == 0);
          unsigned long long a = key[e], c = key[p];
          if (up ? (a > c) : (a < c)) { key[e] = c; key[p] = a; }
        }
      }
    }
  __syncthreads();
  int base = (order * B_ + b) * N_;
  for (int e = tid; e < N_; e += 1024) {
    int orig = (int)(key[e] & 2047ull);
    sidx[base + e]    = orig;
    uidx[base + orig] = e;
  }

  // CDNA5 cluster path: NOP unless dispatched in a workgroup cluster.
  if (__builtin_amdgcn_cluster_id_x() > 0) __builtin_amdgcn_s_cluster_barrier();
  __builtin_amdgcn_s_wait_tensorcnt(0);   // TENSORcnt==0: trivially satisfied
}

// out[b,r,:] = src[b, idx[b,r], :]   (used for both sort-gather and unsort)
__global__ void k_gather(const float* __restrict__ src, const int* __restrict__ idx,
                         float* __restrict__ out) {
  int i = blockIdx.x * blockDim.x + threadIdx.x;   // T_*D_
  if (i >= T_ * D_) return;
  int d = i & (D_ - 1); int r = (i >> 8) & (N_ - 1); int b = i >> 19;
  out[i] = src[((size_t)(b * N_ + idx[b * N_ + r]) << 8) + d];
}

// gamma/beta = cond @ W^T + b
__global__ __launch_bounds__(256) void k_cond(const float* __restrict__ cond,
                                              const float* __restrict__ gw, const float* __restrict__ gb,
                                              const float* __restrict__ bw, const float* __restrict__ bb,
                                              float* __restrict__ gamma, float* __restrict__ beta) {
  int b = blockIdx.x, o = threadIdx.x;
  float ag = gb[o], ab = bb[o];
  for (int k = 0; k < CD_; ++k) {
    float cv = cond[b * CD_ + k];
    ag += gw[o * CD_ + k] * cv;
    ab += bw[o * CD_ + k] * cv;
  }
  gamma[b * D_ + o] = ag; beta[b * D_ + o] = ab;
}

// LayerNorm over D_=256, bf16 output. One block per token.
__global__ __launch_bounds__(256) void k_ln_bf16(const float* __restrict__ x,
                                                 const float* __restrict__ w, const float* __restrict__ bia,
                                                 bhalf* __restrict__ o) {
  __shared__ float s[D_];
  int t = blockIdx.x, d = threadIdx.x;
  float v = x[(size_t)t * D_ + d];
  s[d] = v; __syncthreads();
  for (int off = 128; off > 0; off >>= 1) { if (d < off) s[d] += s[d + off]; __syncthreads(); }
  float mu = s[0] * (1.0f / D_); __syncthreads();
  float c = v - mu;
  s[d] = c * c; __syncthreads();
  for (int off = 128; off > 0; off >>= 1) { if (d < off) s[d] += s[d + off]; __syncthreads(); }
  float r = rsqrtf(s[0] * (1.0f / D_) + 1e-5f);
  o[(size_t)t * D_ + d] = (bhalf)(c * r * w[d] + bia[d]);
}

// ---------------------------------------------------------------------------
// WMMA GEMM:  C[M,N] = A[M,K](bf16,row) @ W[N,K](bf16,row)^T  -> fp32
// 128 threads = 4 waves; each wave owns a 32x32 C tile (2x2 accumulators),
// block covers 64x64.  Fragments double-buffered across K steps so the
// global_load_b128s for step k+32 overlap the 4 WMMAs of step k.
// Fragment layout per CDNA5 ISA 7.12.2 (16-bit A 16x32 / B 32x16):
//   lane<16 : K 0..7 & 16..23 ; lane>=16 : K 8..15 & 24..31
// Requires: M % 64 == 0, N % 64 == 0, K % 32 == 0.
// ---------------------------------------------------------------------------
struct FragLd { v8bf lo, hi; };

__device__ __forceinline__ FragLd ld_frag(const bhalf* p) {
  FragLd f; f.lo = *(const v8bf*)p; f.hi = *(const v8bf*)(p + 16); return f;
}
__device__ __forceinline__ v16bf pack_frag(FragLd f) {
  v16bf r;
#pragma unroll
  for (int i = 0; i < 8; ++i) { r[i] = f.lo[i]; r[8 + i] = f.hi[i]; }
  return r;
}
__device__ __forceinline__ v8f wmma_bf16(v16bf a, v16bf b, v8f c) {
  return __builtin_amdgcn_wmma_f32_16x16x32_bf16(false, a, false, b, (short)0, c, false, false);
}

__global__ __launch_bounds__(128) void k_gemm_bf16(const bhalf* __restrict__ A,
                                                   const bhalf* __restrict__ W,
                                                   float* __restrict__ C,
                                                   int M, int N, int K) {
  const int lane  = threadIdx.x & 31;
  const int wave  = threadIdx.x >> 5;
  const int wm    = wave & 1, wn = wave >> 1;
  const int l15   = lane & 15;
  const int hi16  = (lane < 16) ? 0 : 1;
  const int kbase = hi16 ? 8 : 0;

  const int mrow = blockIdx.y * 64 + wm * 32 + l15;
  const int nrow = blockIdx.x * 64 + wn * 32 + l15;
  const bhalf* Ap0 = A + (size_t)mrow * K + kbase;
  const bhalf* Ap1 = Ap0 + (size_t)16 * K;
  const bhalf* Wp0 = W + (size_t)nrow * K + kbase;
  const bhalf* Wp1 = Wp0 + (size_t)16 * K;

  v8f acc00 = {}, acc01 = {}, acc10 = {}, acc11 = {};
  FragLd fa0 = ld_frag(Ap0), fa1 = ld_frag(Ap1);
  FragLd fb0 = ld_frag(Wp0), fb1 = ld_frag(Wp1);

  for (int k0 = 0; k0 < K; k0 += 32) {
    v16bf a0 = pack_frag(fa0), a1 = pack_frag(fa1);
    v16bf b0 = pack_frag(fb0), b1 = pack_frag(fb1);
    int kn = k0 + 32;
    if (kn < K) {                              // prefetch next frags (overlap WMMA)
      fa0 = ld_frag(Ap0 + kn); fa1 = ld_frag(Ap1 + kn);
      fb0 = ld_frag(Wp0 + kn); fb1 = ld_frag(Wp1 + kn);
      __builtin_prefetch(Ap0 + kn + 32, 0, 3); // global_prefetch_b8 (CDNA5)
      __builtin_prefetch(Wp0 + kn + 32, 0, 3);
    }
    acc00 = wmma_bf16(a0, b0, acc00);
    acc01 = wmma_bf16(a0, b1, acc01);
    acc10 = wmma_bf16(a1, b0, acc10);
    acc11 = wmma_bf16(a1, b1, acc11);
  }

  // C layout: VGPR r -> M = r + (lane<16 ? 0 : 8), N = lane&15
  const int mst = blockIdx.y * 64 + wm * 32 + hi16 * 8;
  const int nst = blockIdx.x * 64 + wn * 32 + l15;
  float* Cp0 = C + (size_t)mst * N + nst;
  float* Cp1 = Cp0 + (size_t)16 * N;
#pragma unroll
  for (int r = 0; r < 8; ++r) {
    Cp0[(size_t)r * N]      = acc00[r];
    Cp0[(size_t)r * N + 16] = acc01[r];
    Cp1[(size_t)r * N]      = acc10[r];
    Cp1[(size_t)r * N + 16] = acc11[r];
  }
}

// causal depthwise conv1d (d_conv=4) + SiLU, logical(scan-direction) order out
__global__ void k_conv_silu(const float* __restrict__ xz,
                            const float* __restrict__ cw, const float* __restrict__ cb,
                            float* __restrict__ xc, bhalf* __restrict__ xcbf, int rev) {
  int i = blockIdx.x * blockDim.x + threadIdx.x;   // B_*N_*DI_
  if (i >= B_ * N_ * DI_) return;
  int d = i & (DI_ - 1); int tau = (i >> 9) & (N_ - 1); int b = i >> 20;
  float acc = cb[d];
#pragma unroll
  for (int j = 0; j < 4; ++j) {
    int tp = tau - 3 + j;
    if (tp >= 0) {
      int phys = rev ? (N_ - 1 - tp) : tp;
      acc += cw[d * 4 + j] * xz[((size_t)(b * N_ + phys) << 10) + d];
    }
  }
  float sv = acc / (1.0f + __expf(-acc));
  xc[i] = sv; xcbf[i] = (bhalf)sv;
}

// dt = softplus(x_dbl[:, :16] @ dt_proj_w^T + b)   (K=16 -> VALU GEMV)
__global__ void k_dt(const float* __restrict__ xdbl,
                     const float* __restrict__ dtw, const float* __restrict__ dtb,
                     float* __restrict__ dt) {
  int i = blockIdx.x * blockDim.x + threadIdx.x;   // T_*DI_
  if (i >= T_ * DI_) return;
  int d = i & (DI_ - 1); int t = i >> 9;
  float s = dtb[d];
#pragma unroll
  for (int r = 0; r < DS_; ++r) s += xdbl[(size_t)t * XDP_ + r] * dtw[d * DS_ + r];
  dt[i] = (s > 20.0f) ? s : log1pf(__expf(s));
}

// Selective scan + D-skip + SiLU(z) gate. One block per batch, thread = channel.
__global__ __launch_bounds__(512) void k_scan(const float* __restrict__ dt,
                                              const float* __restrict__ xdbl,
                                              const float* __restrict__ xc,
                                              const float* __restrict__ xz,
                                              const float* __restrict__ A_log,
                                              const float* __restrict__ Dv,
                                              bhalf* __restrict__ ybf, int rev) {
  const int b = blockIdx.x, d = threadIdx.x;
  __shared__ float sBC[32 * 32];
  float A[DS_], h[DS_];
#pragma unroll
  for (int s = 0; s < DS_; ++s) { A[s] = -__expf(A_log[d * DS_ + s]); h[s] = 0.0f; }
  const float Dd = Dv[d];

  for (int t0 = 0; t0 < N_; t0 += 32) {
    __syncthreads();
    for (int idx = threadIdx.x; idx < 1024; idx += 512) {
      int st = idx >> 5, c = idx & 31;
      sBC[idx] = xdbl[((size_t)(b * N_ + t0 + st)) * XDP_ + DS_ + c];
    }
    __syncthreads();
    for (int tt = 0; tt < 32; ++tt) {
      int tau = t0 + tt;
      size_t tok = (size_t)b * N_ + tau;
      float dtv = dt[tok * DI_ + d];
      float xv  = xc[tok * DI_ + d];
      float dx  = dtv * xv;
      const float* Bv = &sBC[tt * 32];
      const float* Cv = Bv + DS_;
      float acc = 0.0f;
#pragma unroll
      for (int s = 0; s < DS_; ++s) {
        h[s] = __expf(dtv * A[s]) * h[s] + dx * Bv[s];
        acc += h[s] * Cv[s];
      }
      float y = acc + xv * Dd;
      int phys = rev ? (N_ - 1 - tau) : tau;
      float z = xz[((size_t)(b * N_ + phys) << 10) + DI_ + d];
      float g = z / (1.0f + __expf(-z));
      ybf[tok * DI_ + d] = (bhalf)(y * g);
    }
  }
}

// xs_next = (init ? xs : xs_next) + gamma*out + beta, logical->physical remap
__global__ void k_accum(float* __restrict__ xs_next, const float* __restrict__ xs,
                        const float* __restrict__ outp,
                        const float* __restrict__ gamma, const float* __restrict__ beta,
                        int rev, int init) {
  int i = blockIdx.x * blockDim.x + threadIdx.x;   // T_*D_
  if (i >= T_ * D_) return;
  int d = i & (D_ - 1); int l = (i >> 8) & (N_ - 1); int b = i >> 19;
  int tau = rev ? (N_ - 1 - l) : l;
  float v = gamma[b * D_ + d] * outp[((size_t)(b * N_ + tau) << 8) + d] + beta[b * D_ + d];
  xs_next[i] = (init ? xs[i] : xs_next[i]) + v;
}

__global__ void k_concat_bf(const float* __restrict__ r0, const float* __restrict__ r1,
                            bhalf* __restrict__ cat) {
  int i = blockIdx.x * blockDim.x + threadIdx.x;   // T_*512
  if (i >= T_ * 512) return;
  int c = i & 511; int t = i >> 9;
  cat[i] = (bhalf)((c < D_) ? r0[(size_t)t * D_ + c] : r1[(size_t)t * D_ + c - D_]);
}

__global__ void k_bias_gelu_bf(const float* __restrict__ h, const float* __restrict__ b,
                               bhalf* __restrict__ o) {
  int i = blockIdx.x * blockDim.x + threadIdx.x;   // T_*512
  if (i >= T_ * 512) return;
  float v = h[i] + b[i & 511];
  o[i] = (bhalf)(0.5f * v * (1.0f + erff(v * 0.70710678118654752f)));
}

// final: x = h2 + b2 ; LayerNorm -> fp32 output
__global__ __launch_bounds__(256) void k_final_ln(const float* __restrict__ h2,
                                                  const float* __restrict__ b2,
                                                  const float* __restrict__ w, const float* __restrict__ bia,
                                                  float* __restrict__ out) {
  __shared__ float s[D_];
  int t = blockIdx.x, d = threadIdx.x;
  float v = h2[(size_t)t * D_ + d] + b2[d];
  s[d] = v; __syncthreads();
  for (int off = 128; off > 0; off >>= 1) { if (d < off) s[d] += s[d + off]; __syncthreads(); }
  float mu = s[0] * (1.0f / D_); __syncthreads();
  float c = v - mu;
  s[d] = c * c; __syncthreads();
  for (int off = 128; off > 0; off >>= 1) { if (d < off) s[d] += s[d + off]; __syncthreads(); }
  float r = rsqrtf(s[0] * (1.0f / D_) + 1e-5f);
  out[(size_t)t * D_ + d] = c * r * w[d] + bia[d];
}

// ---------------------------------------------------------------------------
// host orchestration
// ---------------------------------------------------------------------------
extern "C" void kernel_launch(void* const* d_in, const int* in_sizes, int n_in,
                              void* d_out, int out_size, void* d_ws, size_t ws_size,
                              hipStream_t stream) {
  (void)in_sizes; (void)n_in; (void)out_size; (void)ws_size;

  // jax pytree flatten order (dict keys sorted): x, xyz, cond,
  // fusion{b1,b2,ln_b,ln_w,w1,w2}, then per layer
  // {A_log,D,beta_b,beta_w,conv_b,conv_w,dt_proj_b,dt_proj_w,gamma_b,gamma_w,
  //  in_proj_w,norm_b,norm_w,out_proj_w,x_proj_w}
  const float* x     = (const float*)d_in[0];
  const float* xyz   = (const float*)d_in[1];
  const float* cond  = (const float*)d_in[2];
  const float* f_b1  = (const float*)d_in[3];
  const float* f_b2  = (const float*)d_in[4];
  const float* f_lnb = (const float*)d_in[5];
  const float* f_lnw = (const float*)d_in[6];
  const float* f_w1  = (const float*)d_in[7];
  const float* f_w2  = (const float*)d_in[8];
  auto LP = [&](int l, int off) { return (const float*)d_in[9 + 15 * l + off]; };
  // 0 A_log, 1 D, 2 beta_b, 3 beta_w, 4 conv_b, 5 conv_w, 6 dt_proj_b,
  // 7 dt_proj_w, 8 gamma_b, 9 gamma_w, 10 in_proj_w, 11 norm_b, 12 norm_w,
  // 13 out_proj_w, 14 x_proj_w

  char* base = (char*)d_ws;
  size_t off = 0;
  auto alloc = [&](size_t bytes) -> void* {
    void* r = base + off;
    off += (bytes + 255) & ~(size_t)255;
    return r;
  };

  bhalf* w_in[4]; bhalf* w_xp[4]; bhalf* w_out[4];
  for (int l = 0; l < 4; ++l) {
    w_in[l]  = (bhalf*)alloc((size_t)1024 * 256 * 2);
    w_xp[l]  = (bhalf*)alloc((size_t)XDP_ * DI_ * 2);   // padded 48 -> 64 rows
    w_out[l] = (bhalf*)alloc((size_t)D_ * DI_ * 2);
  }
  bhalf* w1b   = (bhalf*)alloc((size_t)512 * 512 * 2);
  bhalf* w2b   = (bhalf*)alloc((size_t)256 * 512 * 2);
  int*   sidx  = (int*)alloc((size_t)2 * B_ * N_ * 4);
  int*   uidx  = (int*)alloc((size_t)2 * B_ * N_ * 4);
  float* gamma = (float*)alloc((size_t)4 * B_ * D_ * 4);
  float* beta  = (float*)alloc((size_t)4 * B_ * D_ * 4);
  float* res0  = (float*)alloc((size_t)T_ * D_ * 4);
  float* res1  = (float*)alloc((size_t)T_ * D_ * 4);
  float* xsA   = (float*)alloc((size_t)T_ * D_ * 4);
  float* xsB   = (float*)alloc((size_t)T_ * D_ * 4);
  bhalf* xnbf  = (bhalf*)alloc((size_t)T_ * D_ * 2);
  float* xzbuf = (float*)alloc((size_t)T_ * 1024 * 4);
  float* xc    = (float*)alloc((size_t)T_ * DI_ * 4);
  bhalf* xcbf  = (bhalf*)alloc((size_t)T_ * DI_ * 2);   // reused as concat buffer
  float* xdbl  = (float*)alloc((size_t)T_ * XDP_ * 4);  // stride-64 x_dbl
  float* dtb   = (float*)alloc((size_t)T_ * DI_ * 4);   // reused as fusion h1
  bhalf* ybf   = (bhalf*)alloc((size_t)T_ * DI_ * 2);   // reused as h1 bf16
  float* outp  = (float*)alloc((size_t)T_ * D_ * 4);    // reused as fusion h2

  auto cvt = [&](const float* s, bhalf* d, int n_src, int n_dst) {
    k_f32_to_bf16_pad<<<(n_dst + 255) / 256, 256, 0, stream>>>(s, d, n_src, n_dst);
  };
  for (int l = 0; l < 4; ++l) {
    cvt(LP(l, 10), w_in[l], 1024 * 256, 1024 * 256);
    cvt(LP(l, 14), w_xp[l], XD_ * DI_, XDP_ * DI_);     // zero-pad rows 48..63
    cvt(LP(l, 13), w_out[l], D_ * DI_, D_ * DI_);
  }
  cvt(f_w1, w1b, 512 * 512, 512 * 512);
  cvt(f_w2, w2b, 256 * 512, 256 * 512);

  k_sort<<<2 * B_, 1024, 0, stream>>>(xyz, sidx, uidx);
  for (int l = 0; l < 4; ++l)
    k_cond<<<B_, 256, 0, stream>>>(cond, LP(l, 9), LP(l, 8), LP(l, 3), LP(l, 2),
                                   gamma + l * B_ * D_, beta + l * B_ * D_);

  const int TD  = T_ * D_;
  const int TDI = T_ * DI_;
  for (int order = 0; order < 2; ++order) {
    k_gather<<<TD / 256, 256, 0, stream>>>(x, sidx + order * B_ * N_, xsA);
    float* cur = xsA; float* nxt = xsB;
    for (int l = 0; l < 4; ++l) {
      k_ln_bf16<<<T_, 256, 0, stream>>>(cur, LP(l, 12), LP(l, 11), xnbf);
      k_gemm_bf16<<<dim3(1024 / 64, T_ / 64), 128, 0, stream>>>(xnbf, w_in[l], xzbuf, T_, 1024, 256);
      for (int rev = 0; rev < 2; ++rev) {
        k_conv_silu<<<TDI / 256, 256, 0, stream>>>(xzbuf, LP(l, 5), LP(l, 4), xc, xcbf, rev);
        k_gemm_bf16<<<dim3(XDP_ / 64, T_ / 64), 128, 0, stream>>>(xcbf, w_xp[l], xdbl, T_, XDP_, DI_);
        k_dt<<<TDI / 256, 256, 0, stream>>>(xdbl, LP(l, 7), LP(l, 6), dtb);
        k_scan<<<B_, DI_, 0, stream>>>(dtb, xdbl, xc, xzbuf, LP(l, 0), LP(l, 1), ybf, rev);
        k_gemm_bf16<<<dim3(D_ / 64, T_ / 64), 128, 0, stream>>>(ybf, w_out[l], outp, T_, D_, DI_);
        k_accum<<<TD / 256, 256, 0, stream>>>(nxt, cur, outp, gamma + l * B_ * D_,
                                              beta + l * B_ * D_, rev, rev == 0);
      }
      float* t = cur; cur = nxt; nxt = t;
    }
    k_gather<<<TD / 256, 256, 0, stream>>>(cur, uidx + order * B_ * N_,
                                           order == 0 ? res0 : res1);
  }

  // fusion MLP + final layernorm
  k_concat_bf<<<(T_ * 512) / 256, 256, 0, stream>>>(res0, res1, xcbf);
  k_gemm_bf16<<<dim3(512 / 64, T_ / 64), 128, 0, stream>>>(xcbf, w1b, dtb, T_, 512, 512);
  k_bias_gelu_bf<<<(T_ * 512) / 256, 256, 0, stream>>>(dtb, f_b1, ybf);
  k_gemm_bf16<<<dim3(256 / 64, T_ / 64), 128, 0, stream>>>(ybf, w2b, outp, T_, 256, 512);
  k_final_ln<<<T_, 256, 0, stream>>>(outp, f_b2, f_lnw, f_lnb, (float*)d_out);
}
